// AttentionQualifierAggregation_81355270521103
// MI455X (gfx1250) — matrix operand order repaired and checked by reading, not compiled, not run on hardware
//
#include <hip/hip_runtime.h>
#include <hip/hip_bf16.h>

#define NUM_Q 500000
#define NUM_E 250000
#define DIM   256
#define HEADS 4
#define NEG_SLOPE 0.01f
#define EPS 1e-16f

typedef __attribute__((ext_vector_type(16))) __bf16   v16bf;
typedef __attribute__((ext_vector_type(8)))  unsigned v8u;
typedef __attribute__((ext_vector_type(8)))  float    v8f;

// ---- helpers ----
__device__ __forceinline__ unsigned short f2bf(float f) {
  unsigned u = __float_as_uint(f);
  u += 0x7FFFu + ((u >> 16) & 1u);          // round-to-nearest-even
  return (unsigned short)(u >> 16);
}
__device__ __forceinline__ float bf2f(unsigned short h) {
  return __uint_as_float(((unsigned)h) << 16);
}
// order-preserving encoding of float for unsigned atomicMax
__device__ __forceinline__ unsigned enc_ord(float f) {
  unsigned u = __float_as_uint(f);
  return (u & 0x80000000u) ? ~u : (u | 0x80000000u);
}
__device__ __forceinline__ float dec_ord(unsigned u) {
  return __uint_as_float((u & 0x80000000u) ? (u & 0x7FFFFFFFu) : ~u);
}

// ===================================================================
// K0a: one-time f32 -> bf16 hi/lo split (streaming, float4 in, uint2x2 out)
// ===================================================================
__global__ __launch_bounds__(256) void split_kernel(
    const float* __restrict__ X, unsigned short* __restrict__ Hi,
    unsigned short* __restrict__ Lo, long n4) {
  long i4 = (long)blockIdx.x * 256 + threadIdx.x;
  if (i4 >= n4) return;
  long i = i4 * 4;
  const float4 f = *(const float4*)(X + i);
  float fs[4] = {f.x, f.y, f.z, f.w};
  unsigned short h[4], l[4];
  #pragma unroll
  for (int q = 0; q < 4; ++q) { h[q] = f2bf(fs[q]); l[q] = f2bf(fs[q] - bf2f(h[q])); }
  uint2 hp, lp;
  hp.x = (unsigned)h[0] | ((unsigned)h[1] << 16);
  hp.y = (unsigned)h[2] | ((unsigned)h[3] << 16);
  lp.x = (unsigned)l[0] | ((unsigned)l[1] << 16);
  lp.y = (unsigned)l[2] | ((unsigned)l[3] << 16);
  *(uint2*)(Hi + i) = hp;
  *(uint2*)(Lo + i) = lp;
}

// ===================================================================
// K0b: split + transpose w_q: WT[n][k] = split(w_q[k][n])  (256x256, tiny)
// ===================================================================
__global__ __launch_bounds__(256) void splitT_kernel(
    const float* __restrict__ W, unsigned short* __restrict__ HiT,
    unsigned short* __restrict__ LoT) {
  int t = blockIdx.x * 256 + threadIdx.x;   // 0..65535
  int n = t >> 8;
  int k = t & 255;
  float f = W[k * DIM + n];
  unsigned short h = f2bf(f);
  HiT[n * DIM + k] = h;
  LoT[n * DIM + k] = f2bf(f - bf2f(h));
}

// ===================================================================
// K1: XQ = x_q @ w_q via 3-term bf16 split on v_wmma_f32_16x16x32_bf16.
// Inputs pre-split to bf16 hi/lo (A row-major, B transposed [n][k]).
// Block = 256 threads = 8 waves; tile = 128(M) x 16(N); K staged x32 with
// LDS ping-pong (prefetch regs live only load->ds_store within a stage,
// so no scratch spills; one barrier per stage covers both hazards).
// Grid = (Ntiles=16, Mblocks): N-blocks sharing an A tile are launch-
// adjacent -> x_q hi/lo read from HBM ~once, re-reads served by 192MB L2.
// LDS pitches 16B-aligned & bank-conflict-free -> b128 DS traffic only.
// ===================================================================
__global__ __launch_bounds__(256) void gemm_xq_wmma(
    const unsigned short* __restrict__ AHi, const unsigned short* __restrict__ ALo,
    const unsigned short* __restrict__ BHiT, const unsigned short* __restrict__ BLoT,
    float* __restrict__ C) {
  __shared__ unsigned short aHi[2][128][40];   // 32 k + 8 pad (80B pitch)
  __shared__ unsigned short aLo[2][128][40];
  __shared__ unsigned short bHi[16][264];      // 256 k + 8 pad (528B pitch)
  __shared__ unsigned short bLo[16][264];

  const int tid  = threadIdx.x;
  const int lane = tid & 31;
  const int wave = tid >> 5;
  const int half = lane >> 4;     // K-half selector per ISA layout
  const int mn   = lane & 15;     // M (A frag) / N (B frag, C tile)
  const int  n0  = blockIdx.x * 16;
  const long m0  = (long)blockIdx.y * 128;

  // per-thread A staging coords: tile 128 rows x 32 ushorts (4 uint4/row),
  // 512 uint4 per array -> 2 per thread (rows ar and ar+64)
  const int ar = tid >> 2;
  const int ac = (tid & 3) * 8;
  long rowA0 = m0 + ar;      if (rowA0 >= NUM_Q) rowA0 = NUM_Q - 1;  // clamp-pad
  long rowA1 = m0 + ar + 64; if (rowA1 >= NUM_Q) rowA1 = NUM_Q - 1;

  // --- stage full B panel [n][k] once (straight copy of transposed split) ---
  #pragma unroll
  for (int v = 0; v < 2; ++v) {
    int i4 = tid + v * 256;         // 512 uint4 per array
    int n  = i4 >> 5;               // 32 uint4 per row
    int c  = (i4 & 31) * 8;         // ushort offset
    *(uint4*)&bHi[n][c] = *(const uint4*)&BHiT[(long)(n0 + n) * DIM + c];
    *(uint4*)&bLo[n][c] = *(const uint4*)&BLoT[(long)(n0 + n) * DIM + c];
  }

  // --- stage 0 A: global -> regs -> LDS buf 0 ---
  {
    uint4 h0 = *(const uint4*)&AHi[rowA0 * DIM + ac];
    uint4 h1 = *(const uint4*)&AHi[rowA1 * DIM + ac];
    uint4 l0 = *(const uint4*)&ALo[rowA0 * DIM + ac];
    uint4 l1 = *(const uint4*)&ALo[rowA1 * DIM + ac];
    *(uint4*)&aHi[0][ar][ac]      = h0;
    *(uint4*)&aHi[0][ar + 64][ac] = h1;
    *(uint4*)&aLo[0][ar][ac]      = l0;
    *(uint4*)&aLo[0][ar + 64][ac] = l1;
  }
  __syncthreads();

  v8f acc = {};
  const int arow = wave * 16 + mn;

  #pragma unroll
  for (int st = 0; st < 8; ++st) {
    const int p  = st & 1;
    const int kk = st * 32;

    // prefetch next stage (short register lifetime: load -> ds_store below)
    uint4 h0 = {}, h1 = {}, l0 = {}, l1 = {};
    if (st < 7) {
      h0 = *(const uint4*)&AHi[rowA0 * DIM + kk + 32 + ac];
      h1 = *(const uint4*)&AHi[rowA1 * DIM + kk + 32 + ac];
      l0 = *(const uint4*)&ALo[rowA0 * DIM + kk + 32 + ac];
      l1 = *(const uint4*)&ALo[rowA1 * DIM + kk + 32 + ac];
    }

    // fragments from buf p, per CDNA5 16-bit WMMA layouts:
    // A 16x32: pairs i<4 -> K = 2i + 8*half; i>=4 -> K = 2i+8+8*half
    //   => two contiguous 16B chunks at [arow][8*half], [arow][16+8*half]
    // B 32x16: same K order as A => 16B chunks at [mn][kk+8*half], [mn][kk+16+8*half]
    uint4 a0 = *(const uint4*)&aHi[p][arow][8 * half];
    uint4 a1 = *(const uint4*)&aHi[p][arow][16 + 8 * half];
    uint4 c0 = *(const uint4*)&aLo[p][arow][8 * half];
    uint4 c1 = *(const uint4*)&aLo[p][arow][16 + 8 * half];
    uint4 b0 = *(const uint4*)&bHi[mn][kk + 8 * half];
    uint4 b1 = *(const uint4*)&bHi[mn][kk + 16 + 8 * half];
    uint4 d0 = *(const uint4*)&bLo[mn][kk + 8 * half];
    uint4 d1 = *(const uint4*)&bLo[mn][kk + 16 + 8 * half];
    v8u ahv = {a0.x, a0.y, a0.z, a0.w, a1.x, a1.y, a1.z, a1.w};
    v8u alv = {c0.x, c0.y, c0.z, c0.w, c1.x, c1.y, c1.z, c1.w};
    v8u bhv = {b0.x, b0.y, b0.z, b0.w, b1.x, b1.y, b1.z, b1.w};
    v8u blv = {d0.x, d0.y, d0.z, d0.w, d1.x, d1.y, d1.z, d1.w};
    v16bf AH = __builtin_bit_cast(v16bf, ahv);
    v16bf AL = __builtin_bit_cast(v16bf, alv);
    v16bf BH = __builtin_bit_cast(v16bf, bhv);
    v16bf BL = __builtin_bit_cast(v16bf, blv);

    // 3-term split: hi*hi + hi*lo + lo*hi (~fp32 precision, bf16 rate)
    acc = __builtin_amdgcn_wmma_f32_16x16x32_bf16(false, AH, false, BH, (short)0, acc, false, false);
    acc = __builtin_amdgcn_wmma_f32_16x16x32_bf16(false, AH, false, BL, (short)0, acc, false, false);
    acc = __builtin_amdgcn_wmma_f32_16x16x32_bf16(false, AL, false, BH, (short)0, acc, false, false);

    // commit prefetched stage into the inactive buffer
    if (st < 7) {
      *(uint4*)&aHi[p ^ 1][ar][ac]      = h0;
      *(uint4*)&aHi[p ^ 1][ar + 64][ac] = h1;
      *(uint4*)&aLo[p ^ 1][ar][ac]      = l0;
      *(uint4*)&aLo[p ^ 1][ar + 64][ac] = l1;
    }
    __syncthreads();   // writes to p^1 done; reads of p done
  }

  // C/D layout: VGPR r -> M = r + 8*half, N = lane%16.
  // NUM_Q % 16 == 0 -> single wave-uniform guard, unconditional stores.
  if (m0 + wave * 16 < NUM_Q) {
    #pragma unroll
    for (int r = 0; r < 8; ++r) {
      long row = m0 + wave * 16 + half * 8 + r;
      C[row * DIM + n0 + mn] = acc[r];
    }
  }
}

// ===================================================================
// K2: per-row attention logit + leaky-relu + segment max (atomicMax on
// order-preserving uint). One wave per row, 8 rows per wave.
// ===================================================================
__global__ __launch_bounds__(256) void alpha_kernel(
    const float* __restrict__ XQ, const float* __restrict__ XE,
    const float* __restrict__ W, const int* __restrict__ eid,
    float* __restrict__ alpha, unsigned* __restrict__ segmax) {
  const int lane = threadIdx.x & 31;
  const int wv   = threadIdx.x >> 5;
  const long base = ((long)blockIdx.x * 8 + wv) * 8;

  // weight regs: chunk k covers columns c = lane + 32k, head = k/2
  float wxe[8], wxq[8];
  #pragma unroll
  for (int k = 0; k < 8; ++k) {
    int h = k >> 1;
    int dd = lane + 32 * (k & 1);
    wxe[k] = W[h * 128 + dd];        // weight[h, 0:64]  -> xe part
    wxq[k] = W[h * 128 + 64 + dd];   // weight[h, 64:128]-> xq part
  }

  for (int r = 0; r < 8; ++r) {
    long e = base + r;
    if (e >= NUM_Q) return;          // wave-uniform
    int id = eid[e];
    const float* xq = XQ + e * DIM;
    const float* xe = XE + (long)id * DIM;
    float p[4] = {0.f, 0.f, 0.f, 0.f};
    #pragma unroll
    for (int k = 0; k < 8; ++k) {
      int c = lane + 32 * k;
      p[k >> 1] += xe[c] * wxe[k] + xq[c] * wxq[k];
    }
    #pragma unroll
    for (int s = 16; s > 0; s >>= 1) {   // wave32 butterfly
      p[0] += __shfl_xor(p[0], s);
      p[1] += __shfl_xor(p[1], s);
      p[2] += __shfl_xor(p[2], s);
      p[3] += __shfl_xor(p[3], s);
    }
    if (lane < 4) {
      float a = p[lane];
      a = a > 0.f ? a : NEG_SLOPE * a;
      alpha[e * HEADS + lane] = a;
      atomicMax(&segmax[(long)id * HEADS + lane], enc_ord(a));
    }
  }
}

// ===================================================================
// K3: ex = exp(alpha - segmax[edge]); alpha <- ex; segsum += ex
// ===================================================================
__global__ __launch_bounds__(256) void exp_sum_kernel(
    float* __restrict__ alpha, const int* __restrict__ eid,
    const unsigned* __restrict__ segmax, float* __restrict__ segsum) {
  long t = (long)blockIdx.x * 256 + threadIdx.x;
  if (t >= (long)NUM_Q * HEADS) return;
  long e = t >> 2;
  int  h = (int)(t & 3);
  int id = eid[e];
  float mx = dec_ord(segmax[(long)id * HEADS + h]);
  float ex = __expf(alpha[t] - mx);    // arg <= 0, result in (0,1]
  alpha[t] = ex;
  unsafeAtomicAdd(&segsum[(long)id * HEADS + h], ex);
}

// ===================================================================
// K4: out[edge] += (ex / (segsum+eps)) * XQ   (weighted scatter-add)
// ===================================================================
__global__ __launch_bounds__(256) void scatter_kernel(
    const float* __restrict__ XQ, const float* __restrict__ alpha,
    const float* __restrict__ segsum, const int* __restrict__ eid,
    float* __restrict__ out) {
  const int lane = threadIdx.x & 31;
  const int wv   = threadIdx.x >> 5;
  const long base = ((long)blockIdx.x * 8 + wv) * 8;
  for (int r = 0; r < 8; ++r) {
    long e = base + r;
    if (e >= NUM_Q) return;
    int id = eid[e];
    float coef[4];
    #pragma unroll
    for (int h = 0; h < 4; ++h)
      coef[h] = alpha[e * HEADS + h] / (segsum[(long)id * HEADS + h] + EPS);
    const float* xq = XQ + e * DIM;
    float* o = out + (long)id * DIM;
    #pragma unroll
    for (int k = 0; k < 8; ++k) {
      int c = lane + 32 * k;
      unsafeAtomicAdd(&o[c], coef[k >> 1] * xq[c]);
    }
  }
}

// ===================================================================
extern "C" void kernel_launch(void* const* d_in, const int* in_sizes, int n_in,
                              void* d_out, int out_size, void* d_ws, size_t ws_size,
                              hipStream_t stream) {
  const float* x_q    = (const float*)d_in[0];
  const float* x_edge = (const float*)d_in[1];
  const float* w_q    = (const float*)d_in[2];
  const float* weight = (const float*)d_in[3];
  const int*   eids   = (const int*)d_in[4];
  float* out = (float*)d_out;

  char* ws = (char*)d_ws;
  size_t off = 0;
  float* XQ = (float*)(ws + off);              off += (size_t)NUM_Q * DIM * sizeof(float);      // 512 MB
  unsigned short* AHi = (unsigned short*)(ws + off); off += (size_t)NUM_Q * DIM * 2;            // 256 MB
  unsigned short* ALo = (unsigned short*)(ws + off); off += (size_t)NUM_Q * DIM * 2;            // 256 MB
  unsigned short* WHiT = (unsigned short*)(ws + off); off += (size_t)DIM * DIM * 2;             // 128 KB
  unsigned short* WLoT = (unsigned short*)(ws + off); off += (size_t)DIM * DIM * 2;             // 128 KB
  float* alpha = (float*)(ws + off);           off += (size_t)NUM_Q * HEADS * sizeof(float);    // 8 MB
  unsigned* segmax = (unsigned*)(ws + off);    off += (size_t)NUM_E * HEADS * sizeof(unsigned); // 4 MB
  float* segsum = (float*)(ws + off);

  hipMemsetAsync(out,    0, (size_t)out_size * sizeof(float), stream);
  hipMemsetAsync(segmax, 0, (size_t)NUM_E * HEADS * sizeof(unsigned), stream);
  hipMemsetAsync(segsum, 0, (size_t)NUM_E * HEADS * sizeof(float), stream);

  // K0: one-time bf16 hi/lo splits
  long n4 = (long)NUM_Q * DIM / 4;
  split_kernel<<<(int)((n4 + 255) / 256), 256, 0, stream>>>(x_q, AHi, ALo, n4);
  splitT_kernel<<<(DIM * DIM) / 256, 256, 0, stream>>>(w_q, WHiT, WLoT);

  // K1: GEMM. Grid: N-tiles fastest (x), M-blocks in y.
  dim3 g1(DIM / 16, (31250 + 7) / 8);
  gemm_xq_wmma<<<g1, 256, 0, stream>>>(AHi, ALo, WHiT, WLoT, XQ);

  // K2: 64 rows per block (8 waves x 8 rows)
  int g2 = (NUM_Q + 63) / 64;
  alpha_kernel<<<g2, 256, 0, stream>>>(XQ, x_edge, weight, eids, alpha, segmax);

  // K3: one thread per (row, head)
  long na = (long)NUM_Q * HEADS;
  exp_sum_kernel<<<(int)((na + 255) / 256), 256, 0, stream>>>(alpha, eids, segmax, segsum);

  // K4: weighted scatter
  scatter_kernel<<<g2, 256, 0, stream>>>(XQ, alpha, segsum, eids, out);
}
